// EdgeGatedMPNNLayer_66640712565365
// MI455X (gfx1250) — matrix operand order repaired and checked by reading, not compiled, not run on hardware
//
#include <hip/hip_runtime.h>
#include <math.h>

#define ND 96
#define ED 32
#define IN1 (ND + ED)   // 128
#define WAVES 4

typedef __bf16 bf16_t;
typedef bf16_t v16bf __attribute__((ext_vector_type(16)));
typedef float  v8f   __attribute__((ext_vector_type(8)));

union BfTile {
    v16bf v;
    unsigned short s16[16];
    uint4 q[2];
};

__device__ __forceinline__ unsigned short f2bf(float f) {
    unsigned u = __float_as_uint(f);
    u += 0x7FFFu + ((u >> 16) & 1u);   // round-to-nearest-even
    return (unsigned short)(u >> 16);
}

__device__ __forceinline__ float gelu_exact(float x) {
    return 0.5f * x * (1.0f + erff(x * 0.70710678118654752f));
}

__device__ __forceinline__ float sigmoidf_(float x) {
    return 1.0f / (1.0f + expf(-x));
}

// ---------------------------------------------------------------------------
// Prep: zero aggregation buffer, convert x -> bf16
// ---------------------------------------------------------------------------
__global__ void prep_kernel(const float* __restrict__ x, float* __restrict__ agg,
                            unsigned short* __restrict__ xb, int n) {
    int i = blockIdx.x * blockDim.x + threadIdx.x;
    if (i < n) { agg[i] = 0.0f; xb[i] = f2bf(x[i]); }
}

// Pack a row-major f32 weight W[K][Ncols] into per-lane-contiguous bf16 WMMA
// B-tiles: tile (kt,nt) holds 32 lanes x 16 bf16; lane l covers column
// n = nt*16 + (l&15), K-half (l>>4), element j -> K = kt*32 + (l>>4)*8 + (j&7) + (j>>3)*16.
__global__ void pack_w_kernel(const float* __restrict__ W, unsigned short* __restrict__ dst,
                              int K, int Ncols) {
    int t = blockIdx.x * blockDim.x + threadIdx.x;
    int total = K * Ncols;
    if (t >= total) return;
    int NT = Ncols >> 4;
    int tileIdx = t >> 9;          // 512 elements per tile
    int w = t & 511;
    int lane = w >> 4;
    int j = w & 15;
    int kt = tileIdx / NT;
    int nt = tileIdx - kt * NT;
    int n = nt * 16 + (lane & 15);
    int k = kt * 32 + (lane >> 4) * 8 + (j & 7) + (j >> 3) * 16;
    dst[t] = f2bf(W[k * Ncols + n]);
}

__device__ __forceinline__ void load_btile(BfTile& b, const unsigned short* __restrict__ base,
                                           int tileIdx, int lane) {
    const uint4* p = reinterpret_cast<const uint4*>(base + tileIdx * 512 + lane * 16);
    b.q[0] = p[0];
    b.q[1] = p[1];
}

// ---------------------------------------------------------------------------
// Edge kernel: per wave, a tile of 16 edges.
//   edge_input[16x128] -> (msg MLP, gate MLP) -> gated messages -> atomic scatter
// ---------------------------------------------------------------------------
__global__ void __launch_bounds__(32 * WAVES)
edge_kernel(const int* __restrict__ src, const int* __restrict__ dstIdx,
            const float* __restrict__ ea, const unsigned short* __restrict__ xb,
            const unsigned short* __restrict__ mW1p, const float* __restrict__ mb1,
            const unsigned short* __restrict__ mW2p, const float* __restrict__ mb2,
            const unsigned short* __restrict__ gW1p, const float* __restrict__ gb1,
            const float* __restrict__ gW2, const float* __restrict__ gb2,
            float* __restrict__ agg, int numTiles)
{
    __shared__ unsigned short sHid[WAVES][16 * ND];
    __shared__ int sDst[WAVES][16];

    const int lane = threadIdx.x & 31;
    const int wv   = threadIdx.x >> 5;
    const int tile = blockIdx.x * WAVES + wv;
    if (tile >= numTiles) return;           // wave-uniform: EXEC stays all-1 for WMMA
    const int e0 = tile * 16;
    const int m  = lane & 15;               // A-layout row / C-layout column
    const int q  = lane >> 4;
    const int cn = m;

    if (lane < 16) sDst[wv][lane] = dstIdx[e0 + lane];

    // ---- assemble A: 16 x 128 bf16 edge input (x[src] || edge_attr), 4 K-tiles
    BfTile A[4];
    {
        const int s = src[e0 + m];
        const unsigned short* xrow = xb + (long)s * ND;
#pragma unroll
        for (int kt = 0; kt < 3; ++kt) {
            const int k0 = kt * 32 + q * 8;
            A[kt].q[0] = *reinterpret_cast<const uint4*>(xrow + k0);
            A[kt].q[1] = *reinterpret_cast<const uint4*>(xrow + k0 + 16);
        }
        const float* earow = ea + (long)(e0 + m) * ED;
        const int c0 = q * 8;
        float4 f0 = *reinterpret_cast<const float4*>(earow + c0);
        float4 f1 = *reinterpret_cast<const float4*>(earow + c0 + 4);
        float4 f2 = *reinterpret_cast<const float4*>(earow + c0 + 16);
        float4 f3 = *reinterpret_cast<const float4*>(earow + c0 + 20);
        float t0[8] = {f0.x, f0.y, f0.z, f0.w, f1.x, f1.y, f1.z, f1.w};
        float t1[8] = {f2.x, f2.y, f2.z, f2.w, f3.x, f3.y, f3.z, f3.w};
#pragma unroll
        for (int j = 0; j < 8; ++j) {
            A[3].s16[j]     = f2bf(t0[j]);
            A[3].s16[8 + j] = f2bf(t1[j]);
        }
    }

    // ---- message layer 1: [16x128] @ [128x96] -> GELU -> LDS (bf16)
#pragma unroll
    for (int nt = 0; nt < 6; ++nt) {
        v8f acc = {};
#pragma unroll
        for (int kt = 0; kt < 4; ++kt) {
            BfTile b;
            load_btile(b, mW1p, kt * 6 + nt, lane);
            acc = __builtin_amdgcn_wmma_f32_16x16x32_bf16(false, A[kt].v, false, b.v,
                                                          (short)0, acc, false, false);
        }
        const float bias = mb1[nt * 16 + cn];
#pragma unroll
        for (int v = 0; v < 8; ++v) {
            float h = gelu_exact(acc[v] + bias);
            sHid[wv][(v + 8 * q) * ND + nt * 16 + cn] = f2bf(h);
        }
    }

    // ---- gate layer 1 + dot with gW2 (96 -> 1), accumulated in registers
    float p[8] = {0, 0, 0, 0, 0, 0, 0, 0};
#pragma unroll
    for (int nt = 0; nt < 6; ++nt) {
        v8f acc = {};
#pragma unroll
        for (int kt = 0; kt < 4; ++kt) {
            BfTile b;
            load_btile(b, gW1p, kt * 6 + nt, lane);
            acc = __builtin_amdgcn_wmma_f32_16x16x32_bf16(false, A[kt].v, false, b.v,
                                                          (short)0, acc, false, false);
        }
        const float bias = gb1[nt * 16 + cn];
        const float w2   = gW2[nt * 16 + cn];
#pragma unroll
        for (int v = 0; v < 8; ++v)
            p[v] += gelu_exact(acc[v] + bias) * w2;
    }
    // reduce over the 16 columns (lanes within each 16-lane half)
#pragma unroll
    for (int off = 1; off < 16; off <<= 1) {
#pragma unroll
        for (int v = 0; v < 8; ++v) p[v] += __shfl_xor(p[v], off, 32);
    }
    const float gbias = gb2[0];
    float gate[8];
#pragma unroll
    for (int v = 0; v < 8; ++v) gate[v] = sigmoidf_(p[v] + gbias);

    // ---- message layer 2: hidden[16x96] @ [96x96], gate, scatter-add
    BfTile A2[3];
#pragma unroll
    for (int kt = 0; kt < 3; ++kt) {
        const int k0 = kt * 32 + q * 8;
        A2[kt].q[0] = *reinterpret_cast<const uint4*>(&sHid[wv][m * ND + k0]);
        A2[kt].q[1] = *reinterpret_cast<const uint4*>(&sHid[wv][m * ND + k0 + 16]);
    }
    int drow[8];
#pragma unroll
    for (int v = 0; v < 8; ++v) drow[v] = sDst[wv][v + 8 * q];

#pragma unroll
    for (int nt = 0; nt < 6; ++nt) {
        v8f acc = {};
#pragma unroll
        for (int kt = 0; kt < 3; ++kt) {
            BfTile b;
            load_btile(b, mW2p, kt * 6 + nt, lane);
            acc = __builtin_amdgcn_wmma_f32_16x16x32_bf16(false, A2[kt].v, false, b.v,
                                                          (short)0, acc, false, false);
        }
        const float bias = mb2[nt * 16 + cn];
#pragma unroll
        for (int v = 0; v < 8; ++v) {
            float val = (acc[v] + bias) * gate[v];
            atomicAdd(&agg[(long)drow[v] * ND + nt * 16 + cn], val);
        }
    }
}

// ---------------------------------------------------------------------------
// Node kernel: per wave, a tile of 16 nodes.
//   [x || agg] (16x192) -> update MLP -> residual -> LayerNorm -> out
// ---------------------------------------------------------------------------
__global__ void __launch_bounds__(32 * WAVES)
node_kernel(const float* __restrict__ x, const unsigned short* __restrict__ xb,
            const float* __restrict__ agg,
            const unsigned short* __restrict__ uW1p, const float* __restrict__ ub1,
            const unsigned short* __restrict__ uW2p, const float* __restrict__ ub2,
            const float* __restrict__ lng, const float* __restrict__ lnb,
            float* __restrict__ out, int numTiles)
{
    __shared__ unsigned short sHid[WAVES][16 * ND];

    const int lane = threadIdx.x & 31;
    const int wv   = threadIdx.x >> 5;
    const int tile = blockIdx.x * WAVES + wv;
    if (tile >= numTiles) return;
    const int n0 = tile * 16;
    const int m  = lane & 15;
    const int q  = lane >> 4;
    const int cn = m;

    // ---- assemble A: 16 x 192 bf16 (xb row || agg row), 6 K-tiles
    BfTile A[6];
    {
        const unsigned short* xrow = xb + (long)(n0 + m) * ND;
#pragma unroll
        for (int kt = 0; kt < 3; ++kt) {
            const int k0 = kt * 32 + q * 8;
            A[kt].q[0] = *reinterpret_cast<const uint4*>(xrow + k0);
            A[kt].q[1] = *reinterpret_cast<const uint4*>(xrow + k0 + 16);
        }
        const float* arow = agg + (long)(n0 + m) * ND;
#pragma unroll
        for (int kt = 0; kt < 3; ++kt) {
            const int c0 = kt * 32 + q * 8;
            float4 f0 = *reinterpret_cast<const float4*>(arow + c0);
            float4 f1 = *reinterpret_cast<const float4*>(arow + c0 + 4);
            float4 f2 = *reinterpret_cast<const float4*>(arow + c0 + 16);
            float4 f3 = *reinterpret_cast<const float4*>(arow + c0 + 20);
            float t0[8] = {f0.x, f0.y, f0.z, f0.w, f1.x, f1.y, f1.z, f1.w};
            float t1[8] = {f2.x, f2.y, f2.z, f2.w, f3.x, f3.y, f3.z, f3.w};
#pragma unroll
            for (int j = 0; j < 8; ++j) {
                A[3 + kt].s16[j]     = f2bf(t0[j]);
                A[3 + kt].s16[8 + j] = f2bf(t1[j]);
            }
        }
    }

    // ---- update layer 1: [16x192] @ [192x96] -> GELU -> LDS (bf16)
#pragma unroll
    for (int nt = 0; nt < 6; ++nt) {
        v8f acc = {};
#pragma unroll
        for (int kt = 0; kt < 6; ++kt) {
            BfTile b;
            load_btile(b, uW1p, kt * 6 + nt, lane);
            acc = __builtin_amdgcn_wmma_f32_16x16x32_bf16(false, A[kt].v, false, b.v,
                                                          (short)0, acc, false, false);
        }
        const float bias = ub1[nt * 16 + cn];
#pragma unroll
        for (int v = 0; v < 8; ++v) {
            float h = gelu_exact(acc[v] + bias);
            sHid[wv][(v + 8 * q) * ND + nt * 16 + cn] = f2bf(h);
        }
    }

    // ---- update layer 2 + residual
    BfTile A2[3];
#pragma unroll
    for (int kt = 0; kt < 3; ++kt) {
        const int k0 = kt * 32 + q * 8;
        A2[kt].q[0] = *reinterpret_cast<const uint4*>(&sHid[wv][m * ND + k0]);
        A2[kt].q[1] = *reinterpret_cast<const uint4*>(&sHid[wv][m * ND + k0 + 16]);
    }
    float h[6][8];
#pragma unroll
    for (int nt = 0; nt < 6; ++nt) {
        v8f acc = {};
#pragma unroll
        for (int kt = 0; kt < 3; ++kt) {
            BfTile b;
            load_btile(b, uW2p, kt * 6 + nt, lane);
            acc = __builtin_amdgcn_wmma_f32_16x16x32_bf16(false, A2[kt].v, false, b.v,
                                                          (short)0, acc, false, false);
        }
        const float bias = ub2[nt * 16 + cn];
#pragma unroll
        for (int v = 0; v < 8; ++v) {
            const long row = n0 + v + 8 * q;
            h[nt][v] = x[row * ND + nt * 16 + cn] + acc[v] + bias;
        }
    }

    // ---- LayerNorm over the 96 features of each row
    float s1[8] = {0, 0, 0, 0, 0, 0, 0, 0};
    float s2[8] = {0, 0, 0, 0, 0, 0, 0, 0};
#pragma unroll
    for (int nt = 0; nt < 6; ++nt)
#pragma unroll
        for (int v = 0; v < 8; ++v) { s1[v] += h[nt][v]; s2[v] += h[nt][v] * h[nt][v]; }
#pragma unroll
    for (int off = 1; off < 16; off <<= 1) {
#pragma unroll
        for (int v = 0; v < 8; ++v) {
            s1[v] += __shfl_xor(s1[v], off, 32);
            s2[v] += __shfl_xor(s2[v], off, 32);
        }
    }
    float mu[8], rs[8];
#pragma unroll
    for (int v = 0; v < 8; ++v) {
        mu[v] = s1[v] * (1.0f / ND);
        float var = s2[v] * (1.0f / ND) - mu[v] * mu[v];
        rs[v] = rsqrtf(var + 1e-5f);
    }
#pragma unroll
    for (int nt = 0; nt < 6; ++nt) {
        const int col = nt * 16 + cn;
        const float g = lng[col], b = lnb[col];
#pragma unroll
        for (int v = 0; v < 8; ++v) {
            const long row = n0 + v + 8 * q;
            out[row * ND + col] = (h[nt][v] - mu[v]) * rs[v] * g + b;
        }
    }
}

// ---------------------------------------------------------------------------
extern "C" void kernel_launch(void* const* d_in, const int* in_sizes, int n_in,
                              void* d_out, int out_size, void* d_ws, size_t ws_size,
                              hipStream_t stream) {
    const float* x    = (const float*)d_in[0];
    const int*   ei   = (const int*)d_in[1];
    const float* ea   = (const float*)d_in[2];
    const float* mW1  = (const float*)d_in[3];
    const float* mb1  = (const float*)d_in[4];
    const float* mW2  = (const float*)d_in[5];
    const float* mb2  = (const float*)d_in[6];
    const float* gW1  = (const float*)d_in[7];
    const float* gb1  = (const float*)d_in[8];
    const float* gW2  = (const float*)d_in[9];
    const float* gb2  = (const float*)d_in[10];
    const float* uW1  = (const float*)d_in[11];
    const float* ub1  = (const float*)d_in[12];
    const float* uW2  = (const float*)d_in[13];
    const float* ub2  = (const float*)d_in[14];
    const float* lng  = (const float*)d_in[15];
    const float* lnb  = (const float*)d_in[16];

    const int N = in_sizes[0] / ND;
    const int E = in_sizes[1] / 2;
    const int* src = ei;
    const int* dst = ei + E;

    // workspace layout (16-byte aligned slices)
    char* ws = (char*)d_ws;
    float*          agg  = (float*)ws;
    size_t off = (size_t)N * ND * sizeof(float);
    unsigned short* xb   = (unsigned short*)(ws + off);
    off += (size_t)N * ND * sizeof(unsigned short);
    unsigned short* mW1p = (unsigned short*)(ws + off); off += (size_t)IN1 * ND * 2;
    unsigned short* gW1p = (unsigned short*)(ws + off); off += (size_t)IN1 * ND * 2;
    unsigned short* mW2p = (unsigned short*)(ws + off); off += (size_t)ND * ND * 2;
    unsigned short* uW1p = (unsigned short*)(ws + off); off += (size_t)(2 * ND) * ND * 2;
    unsigned short* uW2p = (unsigned short*)(ws + off);

    const int nx = N * ND;
    prep_kernel<<<(nx + 255) / 256, 256, 0, stream>>>(x, agg, xb, nx);
    pack_w_kernel<<<(IN1 * ND + 255) / 256, 256, 0, stream>>>(mW1, mW1p, IN1, ND);
    pack_w_kernel<<<(IN1 * ND + 255) / 256, 256, 0, stream>>>(gW1, gW1p, IN1, ND);
    pack_w_kernel<<<(ND * ND + 255) / 256, 256, 0, stream>>>(mW2, mW2p, ND, ND);
    pack_w_kernel<<<(2 * ND * ND + 255) / 256, 256, 0, stream>>>(uW1, uW1p, 2 * ND, ND);
    pack_w_kernel<<<(ND * ND + 255) / 256, 256, 0, stream>>>(uW2, uW2p, ND, ND);

    const int eTiles = E / 16;
    edge_kernel<<<(eTiles + WAVES - 1) / WAVES, 32 * WAVES, 0, stream>>>(
        src, dst, ea, xb, mW1p, mb1, mW2p, mb2, gW1p, gb1, gW2, gb2, agg, eTiles);

    const int nTiles = N / 16;
    node_kernel<<<(nTiles + WAVES - 1) / WAVES, 32 * WAVES, 0, stream>>>(
        x, xb, agg, uW1p, ub1, uW2p, ub2, lng, lnb, (float*)d_out, nTiles);
}
